// LinearAttention_67920612819105
// MI455X (gfx1250) — matrix operand order repaired and checked by reading, not compiled, not run on hardware
//
#include <hip/hip_runtime.h>
#include <hip/hip_bf16.h>
#include <stdint.h>

// ---------------------------------------------------------------------------
// Linear attention pipeline for MI455X (gfx1250, wave32, WMMA bf16).
// All GEMMs use v_wmma_f32_16x16x32_bf16; f32 accumulation; softmax in f32.
// K-loops are chunked (64 wide): all b128 fragment loads for a chunk issue
// before its 8 WMMAs so the scheduler can overlap loads with matrix math
// (partial s_wait_loadcnt instead of wait-0 before every WMMA).
// ---------------------------------------------------------------------------

#define BB 8
#define NN 4096
#define CC 768
#define HH 12
#define DD 64
#define ROWS (BB * NN)          // 32768
#define QKVC (3 * CC)           // 2304

typedef __attribute__((ext_vector_type(16))) __bf16 v16bf;
typedef __attribute__((ext_vector_type(8)))  float  v8f;

union FragBF {
    v16bf v;
    uint4 q[2];
    uint32_t u[8];
};

static __device__ inline uint16_t f32_bf16(float f) {
    union { float f; uint32_t u; } x;
    x.f = f;
    uint32_t r = x.u + 0x7FFFu + ((x.u >> 16) & 1u);   // round-to-nearest-even
    return (uint16_t)(r >> 16);
}

static __device__ inline uint32_t pk_bf16(float a, float b) {
    return (uint32_t)f32_bf16(a) | ((uint32_t)f32_bf16(b) << 16);
}

static __device__ inline v8f wmma_bf16(v16bf a, v16bf b, v8f c) {
    // D = A(16x32) * B(32x16) + C ; emits v_wmma_f32_16x16x32_bf16
    return __builtin_amdgcn_wmma_f32_16x16x32_bf16(
        false, a, false, b, (short)0, c, false, false);
}

// A fragment (16x32 bf16, row-major source [M][ld]).
// Lane layout: lanes 0-15 m=lane, K=0..15;
// lanes 16-31 m=lane-16, K=16..31. Two contiguous b128 loads.
static __device__ inline v16bf load_a_bf16(const uint16_t* __restrict__ base,
                                           int ld, int m, int k0, int hf) {
    FragBF f;
    const uint16_t* p = base + (size_t)m * ld + k0;
    f.q[0] = *reinterpret_cast<const uint4*>(p + (hf ? 16 : 0));
    f.q[1] = *reinterpret_cast<const uint4*>(p + (hf ? 24 : 8));
    return f.v;
}

// B fragment (32x16 bf16) from transposed storage Bt[n][ld] (= B column-major).
// Lane layout: lanes 0-15 n=lane K=0..15; lanes 16-31 n=lane-16 K=16..31.
static __device__ inline v16bf load_b_bf16(const uint16_t* __restrict__ base,
                                           int ld, int n, int k0, int hf) {
    FragBF f;
    const uint16_t* p = base + (size_t)n * ld + k0 + (hf ? 16 : 0);
    f.q[0] = *reinterpret_cast<const uint4*>(p);
    f.q[1] = *reinterpret_cast<const uint4*>(p + 8);
    return f.v;
}

// ---------------------------------------------------------------------------
// Kernel 0a: streaming convert f32 -> bf16 (8 elements/thread, b128 in/out)
// ---------------------------------------------------------------------------
__global__ __launch_bounds__(256) void k_cvt_bf16(const float* __restrict__ src,
                                                  uint16_t* __restrict__ dst,
                                                  int n8 /* count/8 */) {
    int idx = blockIdx.x * blockDim.x + threadIdx.x;
    if (idx < n8) {
        const float4* s = reinterpret_cast<const float4*>(src) + idx * 2;
        float4 a = s[0], b = s[1];
        uint4 o;
        o.x = pk_bf16(a.x, a.y); o.y = pk_bf16(a.z, a.w);
        o.z = pk_bf16(b.x, b.y); o.w = pk_bf16(b.z, b.w);
        reinterpret_cast<uint4*>(dst)[idx] = o;
    }
}

// ---------------------------------------------------------------------------
// Kernel 0b: transpose + convert f32 weight [R][Cc] -> bf16 [Cc][R]
// ---------------------------------------------------------------------------
__global__ void k_transpose_cvt(const float* __restrict__ src,
                                uint16_t* __restrict__ dst, int R, int Cc) {
    int idx = blockIdx.x * blockDim.x + threadIdx.x;
    if (idx < R * Cc) {
        int r = idx / Cc, c = idx % Cc;
        dst[(size_t)c * R + r] = f32_bf16(src[idx]);
    }
}

// ---------------------------------------------------------------------------
// Kernel 1: qkv = x @ Wqkv, fused per-head softmax on q,k.
// grid = (36 column tiles = (mat,h), 512 row tiles), block = 128 (4 waves).
// Each block computes a 64x64 tile (rows = flattened b*n, cols = one head's
// D=64 slice of q, k or v). Softmax over D is block-local via an LDS tile.
// q  -> [B,H,N,D] bf16 ; k,v -> transposed [B,H,D,N] bf16.
// ---------------------------------------------------------------------------
__global__ __launch_bounds__(128) void k_qkv(
    const uint16_t* __restrict__ xb, const uint16_t* __restrict__ WqkvT,
    uint16_t* __restrict__ qbuf, uint16_t* __restrict__ kT,
    uint16_t* __restrict__ vT) {
    __shared__ float tile[64 * 65];

    int ct  = blockIdx.x;            // 0..35 : (mat, head)
    int mt  = blockIdx.y;            // 0..511
    int mat = ct / HH;               // 0=q 1=k 2=v
    int h   = ct % HH;
    int wv  = threadIdx.x >> 5;
    int lane = threadIdx.x & 31;
    int ln  = lane & 15;
    int hf  = lane >> 4;
    int mi  = wv;                                 // wave owns 16-row strip
    int mrow = mt * 64 + mi * 16 + ln;            // global row (b*N+n)
    int ncol = ct * 64;                           // global column base

    v8f acc0 = {}, acc1 = {}, acc2 = {}, acc3 = {};
    for (int k0 = 0; k0 < CC; k0 += 64) {
        // Issue ALL 20 b128 loads for this 64-wide K chunk first.
        v16bf aL  = load_a_bf16(xb, CC, mrow, k0, hf);
        v16bf aH  = load_a_bf16(xb, CC, mrow, k0 + 32, hf);
        v16bf b0L = load_b_bf16(WqkvT, CC, ncol + 0  + ln, k0, hf);
        v16bf b1L = load_b_bf16(WqkvT, CC, ncol + 16 + ln, k0, hf);
        v16bf b2L = load_b_bf16(WqkvT, CC, ncol + 32 + ln, k0, hf);
        v16bf b3L = load_b_bf16(WqkvT, CC, ncol + 48 + ln, k0, hf);
        v16bf b0H = load_b_bf16(WqkvT, CC, ncol + 0  + ln, k0 + 32, hf);
        v16bf b1H = load_b_bf16(WqkvT, CC, ncol + 16 + ln, k0 + 32, hf);
        v16bf b2H = load_b_bf16(WqkvT, CC, ncol + 32 + ln, k0 + 32, hf);
        v16bf b3H = load_b_bf16(WqkvT, CC, ncol + 48 + ln, k0 + 32, hf);
        acc0 = wmma_bf16(aL, b0L, acc0);
        acc1 = wmma_bf16(aL, b1L, acc1);
        acc2 = wmma_bf16(aL, b2L, acc2);
        acc3 = wmma_bf16(aL, b3L, acc3);
        acc0 = wmma_bf16(aH, b0H, acc0);
        acc1 = wmma_bf16(aH, b1H, acc1);
        acc2 = wmma_bf16(aH, b2H, acc2);
        acc3 = wmma_bf16(aH, b3H, acc3);
    }

    // Spill 64x64 f32 tile to LDS (C/D layout: lane half -> m += 8).
#pragma unroll
    for (int r = 0; r < 8; r++) {
        int row = mi * 16 + r + hf * 8;
        tile[row * 65 + 0  + ln] = acc0[r];
        tile[row * 65 + 16 + ln] = acc1[r];
        tile[row * 65 + 32 + ln] = acc2[r];
        tile[row * 65 + 48 + ln] = acc3[r];
    }
    __syncthreads();

    // Row softmax over head_dim (only q and k)
    if (mat != 2 && threadIdx.x < 64) {
        float* rp = &tile[threadIdx.x * 65];
        float mx = rp[0];
#pragma unroll 4
        for (int d = 1; d < 64; d++) mx = fmaxf(mx, rp[d]);
        float s = 0.f;
#pragma unroll 4
        for (int d = 0; d < 64; d++) { float e = __expf(rp[d] - mx); rp[d] = e; s += e; }
        float inv = 1.f / s;
#pragma unroll 4
        for (int d = 0; d < 64; d++) rp[d] *= inv;
    }
    __syncthreads();

    int b  = (mt * 64) / NN;
    int n0 = (mt * 64) & (NN - 1);
    size_t bh = (size_t)b * HH + h;

    if (mat == 0) {
        // q: [B,H,N,D] ; packed dword stores (2 bf16 along d)
        uint32_t* dst = reinterpret_cast<uint32_t*>(qbuf + (bh * NN + n0) * DD);
        for (int idx = threadIdx.x; idx < 64 * 32; idx += 128) {
            int nl = idx >> 5, dp = (idx & 31) * 2;
            dst[nl * 32 + (idx & 31)] =
                pk_bf16(tile[nl * 65 + dp], tile[nl * 65 + dp + 1]);
        }
    } else {
        // k,v transposed: [B,H,D,N] ; packed dword stores (2 bf16 along n)
        uint16_t* dstb = (mat == 1 ? kT : vT) + bh * (size_t)DD * NN + n0;
        uint32_t* dst = reinterpret_cast<uint32_t*>(dstb);
        for (int idx = threadIdx.x; idx < 64 * 32; idx += 128) {
            int d = idx >> 5, nl = (idx & 31) * 2;
            dst[d * (NN / 2) + (idx & 31)] =
                pk_bf16(tile[nl * 65 + d], tile[(nl + 1) * 65 + d]);
        }
    }
}

// ---------------------------------------------------------------------------
// Kernel 2: context[d,e] = sum_n k[n,d]*v[n,e] per (b,h); write context^T.
// grid = 96 (b*h), block = 512 (16 waves); wave (mi,ni) owns one 16x16 tile,
// K-loop over N=4096 in 64-wide chunks (128 WMMAs/wave).
// ---------------------------------------------------------------------------
__global__ __launch_bounds__(512) void k_ctx(
    const uint16_t* __restrict__ kT, const uint16_t* __restrict__ vT,
    uint16_t* __restrict__ ctxT) {
    int bh = blockIdx.x;
    int wv = threadIdx.x >> 5;
    int lane = threadIdx.x & 31;
    int ln = lane & 15, hf = lane >> 4;
    int mi = wv >> 2, ni = wv & 3;

    const uint16_t* A  = kT + (size_t)bh * DD * NN;   // [d][n]
    const uint16_t* Bt = vT + (size_t)bh * DD * NN;   // [e][n]
    int m = mi * 16 + ln;   // d
    int e = ni * 16 + ln;   // e

    v8f acc = {};
    for (int k0 = 0; k0 < NN; k0 += 64) {
        v16bf aL = load_a_bf16(A,  NN, m, k0, hf);
        v16bf aH = load_a_bf16(A,  NN, m, k0 + 32, hf);
        v16bf bL = load_b_bf16(Bt, NN, e, k0, hf);
        v16bf bH = load_b_bf16(Bt, NN, e, k0 + 32, hf);
        acc = wmma_bf16(aL, bL, acc);
        acc = wmma_bf16(aH, bH, acc);
    }

    uint16_t* dst = ctxT + (size_t)bh * DD * DD;      // [e][d]
#pragma unroll
    for (int r = 0; r < 8; r++) {
        int d = mi * 16 + r + hf * 8;
        dst[(size_t)e * DD + d] = f32_bf16(acc[r]);
    }
}

// ---------------------------------------------------------------------------
// Kernel 3: out[n,e] = sum_d q[n,d]*context[d,e]; write into [B,N,C] bf16.
// grid = (H, 512 row tiles), block = 128 (4 waves).
// ---------------------------------------------------------------------------
__global__ __launch_bounds__(128) void k_attn(
    const uint16_t* __restrict__ qbuf, const uint16_t* __restrict__ ctxT,
    uint16_t* __restrict__ attn) {
    int h  = blockIdx.x;
    int mt = blockIdx.y;
    int wv = threadIdx.x >> 5;
    int lane = threadIdx.x & 31;
    int ln = lane & 15, hf = lane >> 4;
    int mi = wv;

    int b  = (mt * 64) / NN;
    int n0 = (mt * 64) & (NN - 1);
    size_t bh = (size_t)b * HH + h;

    const uint16_t* A  = qbuf + bh * (size_t)NN * DD; // [n][d]
    const uint16_t* Bt = ctxT + bh * (size_t)DD * DD; // [e][d]
    int m = n0 + mi * 16 + ln;

    // K = 64: single chunk, all loads then all WMMAs.
    v16bf aL  = load_a_bf16(A, DD, m, 0, hf);
    v16bf aH  = load_a_bf16(A, DD, m, 32, hf);
    v16bf b0L = load_b_bf16(Bt, DD, 0  + ln, 0, hf);
    v16bf b1L = load_b_bf16(Bt, DD, 16 + ln, 0, hf);
    v16bf b2L = load_b_bf16(Bt, DD, 32 + ln, 0, hf);
    v16bf b3L = load_b_bf16(Bt, DD, 48 + ln, 0, hf);
    v16bf b0H = load_b_bf16(Bt, DD, 0  + ln, 32, hf);
    v16bf b1H = load_b_bf16(Bt, DD, 16 + ln, 32, hf);
    v16bf b2H = load_b_bf16(Bt, DD, 32 + ln, 32, hf);
    v16bf b3H = load_b_bf16(Bt, DD, 48 + ln, 32, hf);

    v8f acc0 = {}, acc1 = {}, acc2 = {}, acc3 = {};
    acc0 = wmma_bf16(aL, b0L, acc0);
    acc1 = wmma_bf16(aL, b1L, acc1);
    acc2 = wmma_bf16(aL, b2L, acc2);
    acc3 = wmma_bf16(aL, b3L, acc3);
    acc0 = wmma_bf16(aH, b0H, acc0);
    acc1 = wmma_bf16(aH, b1H, acc1);
    acc2 = wmma_bf16(aH, b2H, acc2);
    acc3 = wmma_bf16(aH, b3H, acc3);

#pragma unroll
    for (int r = 0; r < 8; r++) {
        int n = n0 + mi * 16 + r + hf * 8;
        size_t rowbase = ((size_t)b * NN + n) * CC + h * DD;
        attn[rowbase + 0  + ln] = f32_bf16(acc0[r]);
        attn[rowbase + 16 + ln] = f32_bf16(acc1[r]);
        attn[rowbase + 32 + ln] = f32_bf16(acc2[r]);
        attn[rowbase + 48 + ln] = f32_bf16(acc3[r]);
    }
}

// ---------------------------------------------------------------------------
// Kernel 4: Y = attn @ Wproj + bproj -> f32 d_out.
// grid = (12 col tiles, 512 row tiles), block = 128 (4 waves).
// ---------------------------------------------------------------------------
__global__ __launch_bounds__(128) void k_proj(
    const uint16_t* __restrict__ attn, const uint16_t* __restrict__ WprojT,
    const float* __restrict__ bproj, float* __restrict__ out) {
    int ctn = blockIdx.x;    // 0..11
    int mt  = blockIdx.y;    // 0..511
    int wv = threadIdx.x >> 5;
    int lane = threadIdx.x & 31;
    int ln = lane & 15, hf = lane >> 4;
    int mi = wv;

    int m = mt * 64 + mi * 16 + ln;
    int ncol = ctn * 64;

    v8f acc0 = {}, acc1 = {}, acc2 = {}, acc3 = {};
    for (int k0 = 0; k0 < CC; k0 += 64) {
        v16bf aL  = load_a_bf16(attn, CC, m, k0, hf);
        v16bf aH  = load_a_bf16(attn, CC, m, k0 + 32, hf);
        v16bf b0L = load_b_bf16(WprojT, CC, ncol + 0  + ln, k0, hf);
        v16bf b1L = load_b_bf16(WprojT, CC, ncol + 16 + ln, k0, hf);
        v16bf b2L = load_b_bf16(WprojT, CC, ncol + 32 + ln, k0, hf);
        v16bf b3L = load_b_bf16(WprojT, CC, ncol + 48 + ln, k0, hf);
        v16bf b0H = load_b_bf16(WprojT, CC, ncol + 0  + ln, k0 + 32, hf);
        v16bf b1H = load_b_bf16(WprojT, CC, ncol + 16 + ln, k0 + 32, hf);
        v16bf b2H = load_b_bf16(WprojT, CC, ncol + 32 + ln, k0 + 32, hf);
        v16bf b3H = load_b_bf16(WprojT, CC, ncol + 48 + ln, k0 + 32, hf);
        acc0 = wmma_bf16(aL, b0L, acc0);
        acc1 = wmma_bf16(aL, b1L, acc1);
        acc2 = wmma_bf16(aL, b2L, acc2);
        acc3 = wmma_bf16(aL, b3L, acc3);
        acc0 = wmma_bf16(aH, b0H, acc0);
        acc1 = wmma_bf16(aH, b1H, acc1);
        acc2 = wmma_bf16(aH, b2H, acc2);
        acc3 = wmma_bf16(aH, b3H, acc3);
    }

    float bi0 = bproj[ncol + 0  + ln];
    float bi1 = bproj[ncol + 16 + ln];
    float bi2 = bproj[ncol + 32 + ln];
    float bi3 = bproj[ncol + 48 + ln];
#pragma unroll
    for (int r = 0; r < 8; r++) {
        size_t row = (size_t)mt * 64 + mi * 16 + r + hf * 8;
        float* rp = out + row * CC + ncol;
        rp[0  + ln] = acc0[r] + bi0;
        rp[16 + ln] = acc1[r] + bi1;
        rp[32 + ln] = acc2[r] + bi2;
        rp[48 + ln] = acc3[r] + bi3;
    }
}

// ---------------------------------------------------------------------------
extern "C" void kernel_launch(void* const* d_in, const int* in_sizes, int n_in,
                              void* d_out, int out_size, void* d_ws, size_t ws_size,
                              hipStream_t stream) {
    const float* x     = (const float*)d_in[0];   // [B,N,C]
    const float* Wqkv  = (const float*)d_in[1];   // [C,3C]
    const float* Wproj = (const float*)d_in[2];   // [C,C]
    const float* bproj = (const float*)d_in[3];   // [C]
    float* out = (float*)d_out;

    // Carve workspace (bf16 buffers, 256-B aligned)
    uint8_t* ws = (uint8_t*)d_ws;
    size_t off = 0;
    auto carve = [&](size_t bytes) {
        uint8_t* p = ws + off;
        off += (bytes + 255) & ~(size_t)255;
        return p;
    };
    uint16_t* xb     = (uint16_t*)carve((size_t)ROWS * CC * 2);   // x in bf16
    uint16_t* WqkvT  = (uint16_t*)carve((size_t)QKVC * CC * 2);   // [3C][C]
    uint16_t* WprojT = (uint16_t*)carve((size_t)CC * CC * 2);     // [C][C]
    uint16_t* qbuf   = (uint16_t*)carve((size_t)BB * HH * NN * DD * 2);
    uint16_t* kT     = (uint16_t*)carve((size_t)BB * HH * DD * NN * 2);
    uint16_t* vT     = (uint16_t*)carve((size_t)BB * HH * DD * NN * 2);
    uint16_t* ctxT   = (uint16_t*)carve((size_t)BB * HH * DD * DD * 2);
    uint16_t* attn   = (uint16_t*)carve((size_t)BB * NN * CC * 2);
    (void)ws_size; (void)in_sizes; (void)n_in; (void)out_size;

    // 0) x -> bf16 (once), weight transposes + bf16 convert
    {
        int n8 = (ROWS * CC) / 8;
        k_cvt_bf16<<<(n8 + 255) / 256, 256, 0, stream>>>(x, xb, n8);
        int n1 = CC * QKVC;
        k_transpose_cvt<<<(n1 + 255) / 256, 256, 0, stream>>>(Wqkv, WqkvT, CC, QKVC);
        int n2 = CC * CC;
        k_transpose_cvt<<<(n2 + 255) / 256, 256, 0, stream>>>(Wproj, WprojT, CC, CC);
    }
    // 1) qkv GEMM + softmax epilogue
    k_qkv<<<dim3(3 * HH, ROWS / 64), 128, 0, stream>>>(xb, WqkvT, qbuf, kT, vT);
    // 2) context = k^T v
    k_ctx<<<BB * HH, 512, 0, stream>>>(kT, vT, ctxT);
    // 3) out = q @ context -> [B,N,C] bf16
    k_attn<<<dim3(HH, ROWS / 64), 128, 0, stream>>>(qbuf, ctxT, attn);
    // 4) projection + bias -> f32
    k_proj<<<dim3(CC / 64, ROWS / 64), 128, 0, stream>>>(attn, WprojT, bproj, out);
}
